// MNISTModel_56092272886112
// MI455X (gfx1250) — matrix operand order repaired and checked by reading, not compiled, not run on hardware
//
#include <hip/hip_runtime.h>
#include <hip/hip_bf16.h>

// ---------------------------------------------------------------------------
// RealNVP-style flow, 20 coupling layers, B=8192, D=784, H=512, COND=10.
// GEMMs on v_wmma_f32_16x16x32_bf16 (bf16 in / f32 accumulate).
// - Weights repacked per call into WMMA-fragment-major bf16 in d_ws (~53.7MB,
//   L2 resident).
// - 32 rows per WG; each wave computes BOTH 16-row M tiles so every B
//   fragment register load feeds 2 WMMAs (halves L2 weight traffic/FLOP).
// - h state kept in-place in LDS via composed-permutation bookkeeping
//   (h_true[i] == h[cp[i]]), no double buffer.
// ---------------------------------------------------------------------------

typedef __bf16 bf16_t;
typedef __attribute__((ext_vector_type(16))) __bf16 v16bf;
typedef __attribute__((ext_vector_type(8)))  float  v8f;

union FragU { v16bf v; uint4 q[2]; };

constexpr int   B_    = 8192;
constexpr int   D_    = 784;
constexpr int   H_    = 512;
constexpr int   HALF_ = 392;
constexpr int   COND_ = 10;
constexpr int   L_    = 20;
constexpr float CLAMP_C = 0.636f;

constexpr int ROWS  = 32;    // rows per workgroup (two 16-row M tiles)
constexpr int MT    = 2;     // M tiles per wave (B-fragment reuse factor)
constexpr int K1PAD = 416;   // 392 + 10 one-hot, zero-padded to 13*32
constexpr int KT1   = 13;    // GEMM1 K tiles (K=416)
constexpr int NT1   = 32;    // GEMM1 N tiles (N=512)
constexpr int KT2   = 16;    // GEMM2 K tiles (K=512)
constexpr int NT2P  = 56;    // GEMM2 N tiles padded (N=784 -> 896, zeros)
constexpr int NB1   = 4;     // N-tile register blocking, GEMM1 (32/8 waves)
constexpr int NB2   = 7;     // N-tile register blocking, GEMM2 (56/8 waves)

constexpr long FRAG_ELEMS = 512;                        // 32 lanes * 16 bf16
constexpr long W1FRAGS = (long)L_ * 2 * KT1 * NT1;      // 16640
constexpr long W2FRAGS = (long)L_ * 2 * KT2 * NT2P;     // 35840
constexpr long W2OFF   = W1FRAGS * FRAG_ELEMS;          // bf16 elements
constexpr long PACK_TOTAL = (W1FRAGS + W2FRAGS) * FRAG_ELEMS; // ~26.9M bf16

// ---------------------------------------------------------------------------
// Prep: fp32 weights -> fragment-packed bf16.
// B-fragment lane layout (16x16x32 bf16): lane l holds column n = 16*nt +
// (l&15); its 16 values are K = 32*kt + 16*(l>>4) + j, j=0..15, contiguous.
// Out-of-range K (GEMM1, K>=402) and N (GEMM2, n>=784) pack as zero.
// ---------------------------------------------------------------------------
__global__ void pack_weights_kernel(const float* __restrict__ w1s2,
                                    const float* __restrict__ w1s1,
                                    const float* __restrict__ w2s2,
                                    const float* __restrict__ w2s1,
                                    bf16_t* __restrict__ pack) {
  long idx = (long)blockIdx.x * blockDim.x + threadIdx.x;
  if (idx >= PACK_TOTAL) return;
  float v;
  if (idx < W2OFF) {
    long f = idx >> 9; int e = (int)(idx & 511);
    int lane = e >> 4, j = e & 15;
    int nt = (int)(f % NT1); long t = f / NT1;
    int kt = (int)(t % KT1); t /= KT1;
    int s  = (int)(t & 1);
    int l  = (int)(t >> 1);
    int k  = kt * 32 + ((lane >> 4) << 4) + j;
    int n  = nt * 16 + (lane & 15);
    const float* src = s ? w1s1 : w1s2;
    v = (k < HALF_ + COND_) ? src[((long)l * (HALF_ + COND_) + k) * H_ + n] : 0.0f;
  } else {
    long idx2 = idx - W2OFF;
    long f = idx2 >> 9; int e = (int)(idx2 & 511);
    int lane = e >> 4, j = e & 15;
    int nt = (int)(f % NT2P); long t = f / NT2P;
    int kt = (int)(t % KT2); t /= KT2;
    int s  = (int)(t & 1);
    int l  = (int)(t >> 1);
    int k  = kt * 32 + ((lane >> 4) << 4) + j;
    int n  = nt * 16 + (lane & 15);
    const float* src = s ? w2s1 : w2s2;
    v = (n < D_) ? src[((long)l * H_ + k) * D_ + n] : 0.0f;
  }
  pack[idx] = (bf16_t)v;
}

// ---------------------------------------------------------------------------
// Fragment loaders.
// A (16x32 bf16, row-major in LDS, row stride `stride` elems): lane l holds
// row m=l&15; a[0..15] = K 32*kt + 16*(l>>4) + 0..15, contiguous.
// Two contiguous 16B halves -> two ds_load_b128.
// ---------------------------------------------------------------------------
__device__ __forceinline__ v16bf load_afrag_lds(const bf16_t* base, int stride, int kt) {
  const int lane = threadIdx.x & 31;
  const bf16_t* p = base + (lane & 15) * stride + kt * 32 + (lane >> 4) * 16;
  FragU u;
  u.q[0] = *(const uint4*)(p);
  u.q[1] = *(const uint4*)(p + 8);
  return u.v;
}

__device__ __forceinline__ v16bf load_bfrag_gbl(const bf16_t* fragbase) {
  const int lane = threadIdx.x & 31;
  const bf16_t* p = fragbase + lane * 16;   // 32B contiguous per lane
  FragU u;
  u.q[0] = *(const uint4*)(p);
  u.q[1] = *(const uint4*)(p + 8);
  return u.v;
}

__device__ __forceinline__ v8f wmma_bf16(v16bf a, v16bf b, v8f c) {
  return __builtin_amdgcn_wmma_f32_16x16x32_bf16(false, a, false, b,
                                                 (short)0, c, false, false);
}

// ---------------------------------------------------------------------------
// One subnet: mid = relu(A @ W1 + b1) ; r = mid @ W2 + b2
// All tile loops are wave-uniform -> EXEC stays all-ones for WMMA.
// Wave w owns N-tiles {w + 8*b} and both M tiles: each B fragment feeds
// MT=2 WMMAs; each A fragment feeds NB WMMAs.
// ---------------------------------------------------------------------------
__device__ __forceinline__ void subnet_gemms(const bf16_t* __restrict__ pack,
                                             int l, int s,
                                             const float* __restrict__ b1,
                                             const float* __restrict__ b2,
                                             const bf16_t* Abuf,
                                             bf16_t* midbuf,
                                             float* rbuf) {
  const int tid  = threadIdx.x;
  const int wave = tid >> 5;
  const int lane = tid & 31;
  const int half = lane >> 4;
  const int nn   = lane & 15;

  // GEMM1: [32 x 416] x [416 x 512], 4 N-tiles x 2 M-tiles per wave
  {
    const bf16_t* p1 = pack + ((long)((l * 2 + s) * KT1 * NT1) << 9);
    v8f acc[MT][NB1] = {};
#pragma unroll 4
    for (int kt = 0; kt < KT1; ++kt) {
      const v16bf a0 = load_afrag_lds(Abuf, K1PAD, kt);
      const v16bf a1 = load_afrag_lds(Abuf + 16 * K1PAD, K1PAD, kt);
      const bf16_t* prow = p1 + ((long)(kt * NT1 + wave) << 9);
#pragma unroll
      for (int b = 0; b < NB1; ++b) {
        v16bf bf = load_bfrag_gbl(prow + ((long)(8 * b) << 9));
        acc[0][b] = wmma_bf16(a0, bf, acc[0][b]);
        acc[1][b] = wmma_bf16(a1, bf, acc[1][b]);
      }
    }
#pragma unroll
    for (int b = 0; b < NB1; ++b) {
      const int n = (wave + 8 * b) * 16 + nn;
      const float bias = b1[n];
#pragma unroll
      for (int mt = 0; mt < MT; ++mt)
#pragma unroll
        for (int i = 0; i < 8; ++i) {
          float v = acc[mt][b][i] + bias;          // row = mt*16 + i + 8*half
          midbuf[(mt * 16 + i + 8 * half) * H_ + n] = (bf16_t)fmaxf(v, 0.0f);
        }
    }
  }
  __syncthreads();

  // GEMM2: [32 x 512] x [512 x 896(pad)], 7 N-tiles x 2 M-tiles per wave
  {
    const bf16_t* p2 = pack + W2OFF + ((long)((l * 2 + s) * KT2 * NT2P) << 9);
    v8f acc[MT][NB2] = {};
#pragma unroll 2
    for (int kt = 0; kt < KT2; ++kt) {
      const v16bf a0 = load_afrag_lds(midbuf, H_, kt);
      const v16bf a1 = load_afrag_lds(midbuf + 16 * H_, H_, kt);
      const bf16_t* prow = p2 + ((long)(kt * NT2P + wave) << 9);
#pragma unroll
      for (int b = 0; b < NB2; ++b) {
        v16bf bf = load_bfrag_gbl(prow + ((long)(8 * b) << 9));
        acc[0][b] = wmma_bf16(a0, bf, acc[0][b]);
        acc[1][b] = wmma_bf16(a1, bf, acc[1][b]);
      }
    }
#pragma unroll
    for (int b = 0; b < NB2; ++b) {
      const int n = (wave + 8 * b) * 16 + nn;
      if (n < D_) {
        const float bias = b2[n];
#pragma unroll
        for (int mt = 0; mt < MT; ++mt)
#pragma unroll
          for (int i = 0; i < 8; ++i)
            rbuf[(mt * 16 + i + 8 * half) * D_ + n] = acc[mt][b][i] + bias;
      }
    }
  }
  __syncthreads();
}

// ---------------------------------------------------------------------------
// Main persistent kernel: 256 WGs x 256 threads, each WG runs the full
// 20-layer chain for its 32 rows. ~260 KB dynamic LDS.
// Invariant: h_true[m][i] == h[m*784 + cp[i]].
// ---------------------------------------------------------------------------
__global__ void __launch_bounds__(256, 1)
flow_kernel(const float* __restrict__ x,
            const int* __restrict__ labels,
            const int* __restrict__ perms,
            const float* __restrict__ b1s2, const float* __restrict__ b2s2,
            const float* __restrict__ b1s1, const float* __restrict__ b2s1,
            const bf16_t* __restrict__ pack,
            float* __restrict__ out) {
  extern __shared__ char smem[];
  float*  h      = (float*)smem;                  // 32*784 f32 (in-place state)
  float*  rbuf   = h + ROWS * D_;                 // 32*784 f32
  bf16_t* Abuf   = (bf16_t*)(rbuf + ROWS * D_);   // 32*416 bf16 (16B aligned)
  bf16_t* midbuf = Abuf + ROWS * K1PAD;           // 32*512 bf16
  int*    cpA    = (int*)(midbuf + ROWS * H_);    // 784 composed perm
  int*    cpB    = cpA + D_;                      // 784 composed perm (next)
  int*    labL   = cpB + D_;                      // 32
  float*  jacL   = (float*)(labL + ROWS);         // 32

  const int  tid  = threadIdx.x;
  const long row0 = (long)blockIdx.x * ROWS;

  for (int i = tid; i < ROWS * D_; i += 256) h[i] = x[row0 * D_ + i];
  for (int i = tid; i < D_; i += 256) cpA[i] = i;
  if (tid < ROWS) { labL[tid] = labels[row0 + tid]; jacL[tid] = 0.0f; }
  __syncthreads();

  int* cp  = cpA;
  int* cpn = cpB;

  for (int l = 0; l < L_; ++l) {
    // cpn = cp o perm_l  (gather map for this layer's permuted view)
    for (int i = tid; i < D_; i += 256) cpn[i] = cp[perms[l * D_ + i]];
    __syncthreads();

    // A = [x2 (gathered via cpn) | one-hot(label) | zero pad]
    for (int i = tid; i < ROWS * K1PAD; i += 256) {
      int m = i / K1PAD, k = i - m * K1PAD;
      float v = 0.0f;
      if (k < HALF_)              v = h[m * D_ + cpn[HALF_ + k]];
      else if (k < HALF_ + COND_) v = (labL[m] == (k - HALF_)) ? 1.0f : 0.0f;
      Abuf[i] = (bf16_t)v;
    }
    __syncthreads();

    // subnet "s2": r2 = net([x2|c])
    subnet_gemms(pack, l, 0, b1s2 + l * H_, b2s2 + l * D_, Abuf, midbuf, rbuf);

    // y1 = exp(0.636*atan(s2)) * x1 + t2 ; write y1 in-place over x1's slot,
    // restage y1 (bf16) into A for subnet 1.
    for (int i = tid; i < ROWS * HALF_; i += 256) {
      int m = i / HALF_, j = i - m * HALF_;
      float le = CLAMP_C * atanf(rbuf[m * D_ + j]);
      int   slot = cpn[j];
      float x1 = h[m * D_ + slot];
      float y1 = __expf(le) * x1 + rbuf[m * D_ + HALF_ + j];
      h[m * D_ + slot]    = y1;             // same thread, same slot: race-free
      Abuf[m * K1PAD + j] = (bf16_t)y1;     // one-hot cols 392..415 unchanged
      atomicAdd(&jacL[m], le);
    }
    __syncthreads();

    // subnet "s1": r1 = net([y1|c])
    subnet_gemms(pack, l, 1, b1s1 + l * H_, b2s1 + l * D_, Abuf, midbuf, rbuf);

    // y2 = exp(0.636*atan(s1)) * x2 + t1 ; write y2 in-place over x2's slot
    for (int i = tid; i < ROWS * HALF_; i += 256) {
      int m = i / HALF_, j = i - m * HALF_;
      float le = CLAMP_C * atanf(rbuf[m * D_ + j]);
      int   slot = cpn[HALF_ + j];
      float x2 = h[m * D_ + slot];
      float y2 = __expf(le) * x2 + rbuf[m * D_ + HALF_ + j];
      h[m * D_ + slot] = y2;
      atomicAdd(&jacL[m], le);
    }
    __syncthreads();

    int* t = cp; cp = cpn; cpn = t;         // cp <- cp o perm_l
  }

  // De-scramble through the terminal composed permutation.
  for (int i = tid; i < ROWS * D_; i += 256) {
    int m = i / D_, col = i - m * D_;
    out[row0 * D_ + i] = h[m * D_ + cp[col]];
  }
  if (tid < ROWS) out[(long)B_ * D_ + row0 + tid] = jacL[tid];
}

// ---------------------------------------------------------------------------
extern "C" void kernel_launch(void* const* d_in, const int* in_sizes, int n_in,
                              void* d_out, int out_size, void* d_ws, size_t ws_size,
                              hipStream_t stream) {
  const float* x     = (const float*)d_in[0];
  const int*   labels= (const int*)  d_in[1];
  const int*   perms = (const int*)  d_in[2];
  const float* w1s2  = (const float*)d_in[3];
  const float* b1s2  = (const float*)d_in[4];
  const float* w2s2  = (const float*)d_in[5];
  const float* b2s2  = (const float*)d_in[6];
  const float* w1s1  = (const float*)d_in[7];
  const float* b1s1  = (const float*)d_in[8];
  const float* w2s1  = (const float*)d_in[9];
  const float* b2s1  = (const float*)d_in[10];

  bf16_t* pack = (bf16_t*)d_ws;   // needs ~53.7 MB of workspace

  // 1) repack weights to WMMA-fragment bf16 (recomputed every call; stateless)
  {
    const int  threads = 256;
    const long blocks  = (PACK_TOTAL + threads - 1) / threads;
    pack_weights_kernel<<<(int)blocks, threads, 0, stream>>>(w1s2, w1s1, w2s2, w2s1, pack);
  }

  // 2) full flow
  const size_t smem = (size_t)(2 * ROWS * D_) * 4   // h, rbuf
                    + (size_t)(ROWS * K1PAD) * 2    // Abuf
                    + (size_t)(ROWS * H_) * 2       // midbuf
                    + (size_t)(2 * D_) * 4          // cpA, cpB
                    + (size_t)ROWS * 4              // labels
                    + (size_t)ROWS * 4;             // jac
  (void)hipFuncSetAttribute((const void*)flow_kernel,
                            hipFuncAttributeMaxDynamicSharedMemorySize, (int)smem);
  flow_kernel<<<B_ / ROWS, 256, smem, stream>>>(x, labels, perms,
                                                b1s2, b2s2, b1s1, b2s1,
                                                pack, (float*)d_out);
}
